// RPNModule_51281909514895
// MI455X (gfx1250) — matrix operand order repaired and checked by reading, not compiled, not run on hardware
//
#include <hip/hip_runtime.h>
#include <hip/hip_bf16.h>
#include <math.h>

typedef __attribute__((ext_vector_type(16))) _Float16 v16h;
typedef __attribute__((ext_vector_type(8)))  float    v8f;

#define NBINS 4096
#define CAND  6000
#define POSTN 1000
#define LVLS  5
#define NEGV  (-1000000000.0f)
#define BBOX_CLIP 4.135166556742356f   // log(1000/16)

// ---------------------------------------------------------------------------
// Pack conv weights (256,256,3,3) f32 -> f16 B-fragment layout:
// idx = (((tap*8 + ktile)*16 + ntile)*32 + lane)*16 + n
// lane&15 = column (oc within n-tile); element n -> K = 8*(lane>>4) + (n&7) + 16*(n>>3).
// ---------------------------------------------------------------------------
__global__ void pack_w(const float* __restrict__ cw, _Float16* __restrict__ wp) {
    int idx = blockIdx.x * 256 + threadIdx.x;     // 9*8*16*32*16 = 589824 total
    int n     =  idx        & 15;
    int lane  = (idx >> 4)  & 31;
    int ntile = (idx >> 9)  & 15;
    int kt    = (idx >> 13) &  7;
    int tap   =  idx >> 16;
    if (tap >= 9) return;
    int oc = ntile * 16 + (lane & 15);
    int ic = kt * 32 + (lane >> 4) * 8 + (n & 7) + (n >> 3) * 16;
    int kh = tap / 3, kw = tap - kh * 3;
    wp[idx] = (_Float16)cw[((oc * 256 + ic) * 3 + kh) * 3 + kw];
}

// ---------------------------------------------------------------------------
// Fused: 3x3 conv (WMMA implicit GEMM) + bias + ReLU + 1x1 cls/bbox heads +
// sigmoid, writing scores[(h,w,a)] and reg[(h,w,a),4] for this level.
// Block = 256 threads (8 waves); tile = 16 pixels x 256 oc.
// ---------------------------------------------------------------------------
__global__ __launch_bounds__(256) void conv_fused(
    const float* __restrict__ feat, const _Float16* __restrict__ wp,
    const float* __restrict__ convb, const float* __restrict__ clsw,
    const float* __restrict__ clsb,  const float* __restrict__ bbw,
    const float* __restrict__ bbb,
    float* __restrict__ scores, float* __restrict__ reg,
    int H, int W, int lvl_off, int totn)
{
    __shared__ _Float16 inT[3 * 18 * 256];   // [row][col][ic], f16
    __shared__ float    tT[16 * 256];        // [pixel][oc], post-ReLU

    const int tid = threadIdx.x;
    const int pw0 = blockIdx.x * 16, ph = blockIdx.y, b = blockIdx.z;

    const v16h* wpv = (const v16h*)wp;
    const int wave = tid >> 5, lane = tid & 31;

    // Warm L2/L0 for this wave's first-tap B fragments while staging runs.
    __builtin_prefetch((const void*)&wpv[(2 * wave) * 32 + lane], 0, 3);
    __builtin_prefetch((const void*)&wpv[(2 * wave + 1) * 32 + lane], 0, 3);

    // Stage 3x18x256 input patch (zero-padded borders) as f16.
    for (int idx = tid; idx < 3 * 18 * 256; idx += 256) {
        int ic = idx & 255, cc = idx >> 8;
        int r = cc / 18, c = cc - r * 18;
        int gh = ph - 1 + r, gw = pw0 - 1 + c;
        float v = 0.f;
        if (gh >= 0 && gh < H && gw >= 0 && gw < W)
            v = feat[(((size_t)b * 256 + ic) * H + gh) * W + gw];
        inT[idx] = (_Float16)v;
    }
    __syncthreads();

    const int m = lane & 15, hi = lane >> 4;

    v8f acc0 = {0.f,0.f,0.f,0.f,0.f,0.f,0.f,0.f};
    v8f acc1 = {0.f,0.f,0.f,0.f,0.f,0.f,0.f,0.f};

    for (int tap = 0; tap < 9; ++tap) {
        int kh = tap / 3, kw = tap - kh * 3;
        const _Float16* arow = &inT[(kh * 18 + m + kw) * 256];
        // Pull next tap's weight stream toward this WGP ahead of use
        // (emits global_prefetch_b8; weights are L2-resident, this hides
        // the L2->WGP latency behind the current tap's 16 WMMAs).
        if (tap < 8) {
            __builtin_prefetch(
                (const void*)&wpv[(((tap + 1) * 8) * 16 + 2 * wave) * 32 + lane], 0, 3);
            __builtin_prefetch(
                (const void*)&wpv[(((tap + 1) * 8) * 16 + 2 * wave + 1) * 32 + lane], 0, 3);
        }
        for (int kt = 0; kt < 8; ++kt) {
            // A fragment: lane element e -> K = e + (e>=8?8:0) + (lane>=16?8:0)
            union { v16h v; unsigned int u[8]; } a;
            int base = kt * 32 + hi * 8;
#pragma unroll
            for (int ee = 0; ee < 8; ++ee) {
                int e = ee * 2;
                int K = base + e + ((e >= 8) ? 8 : 0);
                a.u[ee] = *(const unsigned int*)&arow[K];   // 2 contiguous f16
            }
            v16h b0 = wpv[((tap * 8 + kt) * 16 + 2 * wave)     * 32 + lane];
            v16h b1 = wpv[((tap * 8 + kt) * 16 + 2 * wave + 1) * 32 + lane];
            acc0 = __builtin_amdgcn_wmma_f32_16x16x32_f16(false, a.v, false, b0,
                                                          (short)0, acc0, false, false);
            acc1 = __builtin_amdgcn_wmma_f32_16x16x32_f16(false, a.v, false, b1,
                                                          (short)0, acc1, false, false);
        }
    }

    // D layout: lanes 0-15 -> M = vgpr, lanes 16-31 -> M = vgpr+8; N = lane&15.
    int nl = lane & 15;
#pragma unroll
    for (int v = 0; v < 8; ++v) {
        int pos = v + hi * 8;
        int oc0 = wave * 32 + nl, oc1 = wave * 32 + 16 + nl;
        tT[pos * 256 + oc0] = fmaxf(acc0[v] + convb[oc0], 0.f);
        tT[pos * 256 + oc1] = fmaxf(acc1[v] + convb[oc1], 0.f);
    }
    __syncthreads();

    // 1x1 heads: thread = (pixel p, channel j). j<16 -> bbox ch j; j<4 also cls.
    int p = tid >> 4, j = tid & 15;
    const float* trow = &tT[p * 256];
    float db = bbb[j];
    for (int oc = 0; oc < 256; ++oc) db += bbw[j * 256 + oc] * trow[oc];
    size_t n0 = (size_t)b * totn + lvl_off + (size_t)(ph * W + pw0 + p) * 4;
    reg[(n0 + (j >> 2)) * 4 + (j & 3)] = db;     // a = j/4, comp = j%4
    if (j < 4) {
        float dc = clsb[j];
        for (int oc = 0; oc < 256; ++oc) dc += clsw[j * 256 + oc] * trow[oc];
        scores[n0 + j] = 1.f / (1.f + expf(-dc));
    }
}

// ---------------------------------------------------------------------------
__global__ void zero_k(int* p, int n) {
    int i = blockIdx.x * 256 + threadIdx.x;
    if (i < n) p[i] = 0;
}

__global__ void hist_k(const float* __restrict__ scores, int lvl_off, int totn,
                       int N, int* __restrict__ hist) {
    int i = blockIdx.x * 256 + threadIdx.x, b = blockIdx.y;
    if (i >= N) return;
    float s = scores[(size_t)b * totn + lvl_off + i];
    int bin = (int)(s * NBINS);
    bin = min(max(bin, 0), NBINS - 1);
    atomicAdd(&hist[b * NBINS + bin], 1);
}

__global__ void thresh_k(const int* __restrict__ hist, int k, int* __restrict__ tinfo) {
    int b = blockIdx.x;
    if (threadIdx.x != 0) return;
    int cum = 0, tb = 0, above = 0, take = k;
    for (int bin = NBINS - 1; bin >= 0; --bin) {
        int c = hist[b * NBINS + bin];
        if (cum + c >= k) { tb = bin; above = cum; take = k - cum; break; }
        cum += c;
    }
    tinfo[b * 4 + 0] = tb; tinfo[b * 4 + 1] = above; tinfo[b * 4 + 2] = take;
}

__global__ void compact_k(const float* __restrict__ scores, int lvl_off, int totn,
                          int N, int k, const int* __restrict__ tinfo,
                          int* __restrict__ cnt, float* __restrict__ topv,
                          int* __restrict__ topi) {
    int i = blockIdx.x * 256 + threadIdx.x, b = blockIdx.y;
    if (i >= N) return;
    float s = scores[(size_t)b * totn + lvl_off + i];
    int bin = min(max((int)(s * NBINS), 0), NBINS - 1);
    int tb = tinfo[b * 4], above = tinfo[b * 4 + 1], take = tinfo[b * 4 + 2];
    int slot = -1;
    if (bin > tb) slot = atomicAdd(&cnt[b * 2], 1);
    else if (bin == tb) {
        int p = atomicAdd(&cnt[b * 2 + 1], 1);
        if (p < take) slot = above + p;
    }
    if (slot >= 0 && slot < k) {
        topv[(size_t)b * CAND + slot] = s;
        topi[(size_t)b * CAND + slot] = i;
    }
}

// ---------------------------------------------------------------------------
__global__ void decode_k(const int* __restrict__ topi, const float* __restrict__ reg,
                         int lvl_off, int totn, int W, float fsize, float fstride,
                         int k, const int* __restrict__ ih, const int* __restrict__ iw,
                         float* __restrict__ boxes, float* __restrict__ areas) {
    int i = blockIdx.x * 256 + threadIdx.x, b = blockIdx.y;
    if (i >= k) return;
    int n = topi[(size_t)b * CAND + i];
    int a = n & 3, pw = (n >> 2) % W, phh = (n >> 2) / W;
    const double R[4] = {0.2323283, 0.63365731, 1.28478321, 3.15089189};
    double area = (double)fsize * (double)fsize;
    double wsd = sqrt(area / R[a]);
    double hsd = wsd * R[a];
    double sx = (double)pw * (double)fstride, sy = (double)phh * (double)fstride;
    float x1a = (float)(sx - wsd * 0.5), y1a = (float)(sy - hsd * 0.5);
    float x2a = (float)(sx + wsd * 0.5), y2a = (float)(sy + hsd * 0.5);
    float wa = x2a - x1a + 1.f, ha = y2a - y1a + 1.f;
    float cxa = x1a + 0.5f * wa, cya = y1a + 0.5f * ha;
    size_t rb = ((size_t)b * totn + lvl_off + n) * 4;
    float dx = reg[rb], dy = reg[rb + 1];
    float dw = fminf(reg[rb + 2], BBOX_CLIP), dh = fminf(reg[rb + 3], BBOX_CLIP);
    float cx = dx * wa + cxa, cy = dy * ha + cya;
    float w2 = expf(dw) * wa, h2 = expf(dh) * ha;
    float bx1 = cx - 0.5f * w2, by1 = cy - 0.5f * h2;
    float bx2 = cx + 0.5f * w2 - 1.f, by2 = cy + 0.5f * h2 - 1.f;
    float W1 = (float)iw[0] - 1.f, H1 = (float)ih[0] - 1.f;
    bx1 = fminf(fmaxf(bx1, 0.f), W1); by1 = fminf(fmaxf(by1, 0.f), H1);
    bx2 = fminf(fmaxf(bx2, 0.f), W1); by2 = fminf(fmaxf(by2, 0.f), H1);
    size_t o = ((size_t)b * CAND + i) * 4;
    boxes[o] = bx1; boxes[o + 1] = by1; boxes[o + 2] = bx2; boxes[o + 3] = by2;
    areas[(size_t)b * CAND + i] = (bx2 - bx1) * (by2 - by1);
}

// ---------------------------------------------------------------------------
// NMS: one block per image; scores+areas in LDS, n_out iterations of
// first-max argmax + IoU suppression (iou > 0.7 -> NEG).
// ---------------------------------------------------------------------------
__global__ __launch_bounds__(256) void nms_k(
    const float* __restrict__ boxes, const float* __restrict__ areas,
    const float* __restrict__ topv, int k, int n_out, int lvl,
    float* __restrict__ catb, float* __restrict__ cats)
{
    __shared__ float s[CAND];
    __shared__ float ar[CAND];
    __shared__ float rv[256];
    __shared__ int   ri[256];
    __shared__ float bb[5];
    __shared__ int   bi;

    int tid = threadIdx.x, b = blockIdx.x;
    const float* bx = boxes + (size_t)b * CAND * 4;
    for (int i = tid; i < k; i += 256) {
        s[i]  = topv[(size_t)b * CAND + i];
        ar[i] = areas[(size_t)b * CAND + i];
    }
    __syncthreads();

    for (int it = 0; it < n_out; ++it) {
        float bv = -3.0e38f; int bid = 0;
        for (int i = tid; i < k; i += 256) {
            float v = s[i];
            if (v > bv) { bv = v; bid = i; }
        }
        rv[tid] = bv; ri[tid] = bid;
        __syncthreads();
        for (int off = 128; off > 0; off >>= 1) {
            if (tid < off) {
                float ov = rv[tid + off]; int oi = ri[tid + off];
                if (ov > rv[tid] || (ov == rv[tid] && oi < ri[tid])) {
                    rv[tid] = ov; ri[tid] = oi;
                }
            }
            __syncthreads();
        }
        if (tid == 0) {
            int ci = ri[0];
            bi = ci; bb[4] = rv[0];
            bb[0] = bx[ci * 4];     bb[1] = bx[ci * 4 + 1];
            bb[2] = bx[ci * 4 + 2]; bb[3] = bx[ci * 4 + 3];
        }
        __syncthreads();
        int ci = bi; float cv = bb[4];
        float cx1 = bb[0], cy1 = bb[1], cx2 = bb[2], cy2 = bb[3];
        float ca = ar[ci];
        bool valid = cv > NEGV * 0.5f;
        for (int i = tid; i < k; i += 256) {
            float xx1 = fmaxf(cx1, bx[i * 4]),     yy1 = fmaxf(cy1, bx[i * 4 + 1]);
            float xx2 = fminf(cx2, bx[i * 4 + 2]), yy2 = fminf(cy2, bx[i * 4 + 3]);
            float inter = fmaxf(xx2 - xx1, 0.f) * fmaxf(yy2 - yy1, 0.f);
            float iou = inter / (ca + ar[i] - inter + 1e-9f);
            if (iou > 0.7f) s[i] = NEGV;
        }
        if (tid == 0) {
            s[ci] = NEGV;
            size_t o = (size_t)b * (LVLS * POSTN) + (size_t)lvl * POSTN + it;
            if (valid) {
                catb[o * 4]     = bb[0]; catb[o * 4 + 1] = bb[1];
                catb[o * 4 + 2] = bb[2]; catb[o * 4 + 3] = bb[3];
                cats[o] = cv;
            } else {  // reference: boxes[max(keep,0)] == boxes[0], score = NEG
                catb[o * 4]     = bx[0]; catb[o * 4 + 1] = bx[1];
                catb[o * 4 + 2] = bx[2]; catb[o * 4 + 3] = bx[3];
                cats[o] = NEGV;
            }
        }
        __syncthreads();
    }
    // Pad to POSTN entries (level 4: 640 -> 1000) with zeros / NEG.
    for (int idx = tid; idx < POSTN - n_out; idx += 256) {
        size_t o = (size_t)b * (LVLS * POSTN) + (size_t)lvl * POSTN + n_out + idx;
        catb[o * 4] = 0.f; catb[o * 4 + 1] = 0.f;
        catb[o * 4 + 2] = 0.f; catb[o * 4 + 3] = 0.f;
        cats[o] = NEGV;
    }
}

// ---------------------------------------------------------------------------
// Final global top-1000 over 5000 concatenated candidates per image.
// ---------------------------------------------------------------------------
__global__ __launch_bounds__(256) void final_k(const float* __restrict__ catb,
                                               const float* __restrict__ cats,
                                               float* __restrict__ out) {
    __shared__ float s[LVLS * POSTN];
    __shared__ float rv[256];
    __shared__ int   ri[256];
    const int M = LVLS * POSTN;
    int tid = threadIdx.x, b = blockIdx.x;
    for (int i = tid; i < M; i += 256) s[i] = cats[(size_t)b * M + i];
    __syncthreads();
    for (int it = 0; it < POSTN; ++it) {
        float bv = -3.0e38f; int bid = 0;
        for (int i = tid; i < M; i += 256) {
            float v = s[i];
            if (v > bv) { bv = v; bid = i; }
        }
        rv[tid] = bv; ri[tid] = bid;
        __syncthreads();
        for (int off = 128; off > 0; off >>= 1) {
            if (tid < off) {
                float ov = rv[tid + off]; int oi = ri[tid + off];
                if (ov > rv[tid] || (ov == rv[tid] && oi < ri[tid])) {
                    rv[tid] = ov; ri[tid] = oi;
                }
            }
            __syncthreads();
        }
        if (tid == 0) {
            int ci = ri[0];
            s[ci] = NEGV;
            size_t o = ((size_t)b * POSTN + it) * 5;
            const float* bp = &catb[((size_t)b * M + ci) * 4];
            out[o] = bp[0]; out[o + 1] = bp[1]; out[o + 2] = bp[2]; out[o + 3] = bp[3];
            out[o + 4] = rv[0];
        }
        __syncthreads();
    }
}

// ---------------------------------------------------------------------------
extern "C" void kernel_launch(void* const* d_in, const int* in_sizes, int n_in,
                              void* d_out, int out_size, void* d_ws, size_t ws_size,
                              hipStream_t stream) {
    (void)in_sizes; (void)n_in; (void)out_size; (void)ws_size;
    const float* feats[5] = {
        (const float*)d_in[0], (const float*)d_in[1], (const float*)d_in[2],
        (const float*)d_in[3], (const float*)d_in[4]};
    const float* conv_w = (const float*)d_in[5];
    const float* conv_b = (const float*)d_in[6];
    const float* cls_w  = (const float*)d_in[7];
    const float* cls_b  = (const float*)d_in[8];
    const float* bbox_w = (const float*)d_in[9];
    const float* bbox_b = (const float*)d_in[10];
    const int*   ih     = (const int*)d_in[11];
    const int*   iw     = (const int*)d_in[12];

    static const int Hs[5]      = {160, 80, 40, 20, 10};
    static const int Ws[5]      = {256, 128, 64, 32, 16};
    static const int sizesA[5]  = {32, 64, 128, 256, 512};
    static const int stridesA[5]= {4, 8, 16, 32, 64};
    static const int lvl_off[5] = {0, 163840, 204800, 215040, 217600};
    static const int kArr[5]    = {6000, 6000, 6000, 2560, 640};
    static const int noutArr[5] = {1000, 1000, 1000, 1000, 640};
    const int TOTN = 218240, Bn = 2;

    char* ws = (char*)d_ws;
    size_t off = 0;
    auto carve = [&](size_t bytes) -> char* {
        char* p = ws + off;
        off += (bytes + 255) & ~(size_t)255;
        return p;
    };
    _Float16* wp   = (_Float16*)carve((size_t)589824 * 2);
    float* scores  = (float*)carve((size_t)Bn * TOTN * 4);
    float* reg     = (float*)carve((size_t)Bn * TOTN * 16);
    int*   hist    = (int*)  carve((size_t)Bn * NBINS * 4);
    int*   cnt     = (int*)  carve((size_t)Bn * 2 * 4);
    int*   tinfo   = (int*)  carve((size_t)Bn * 4 * 4);
    float* topv    = (float*)carve((size_t)LVLS * Bn * CAND * 4);
    int*   topi    = (int*)  carve((size_t)LVLS * Bn * CAND * 4);
    float* boxes   = (float*)carve((size_t)LVLS * Bn * CAND * 16);
    float* areas   = (float*)carve((size_t)LVLS * Bn * CAND * 4);
    float* catb    = (float*)carve((size_t)Bn * LVLS * POSTN * 16);
    float* cats    = (float*)carve((size_t)Bn * LVLS * POSTN * 4);
    float* outf    = (float*)d_out;

    pack_w<<<589824 / 256, 256, 0, stream>>>(conv_w, wp);

    for (int l = 0; l < LVLS; ++l) {
        conv_fused<<<dim3(Ws[l] / 16, Hs[l], Bn), 256, 0, stream>>>(
            feats[l], wp, conv_b, cls_w, cls_b, bbox_w, bbox_b,
            scores, reg, Hs[l], Ws[l], lvl_off[l], TOTN);
    }

    for (int l = 0; l < LVLS; ++l) {
        const int N = Hs[l] * Ws[l] * 4;
        const int k = kArr[l], n_out = noutArr[l];
        float* topv_l  = topv  + (size_t)l * Bn * CAND;
        int*   topi_l  = topi  + (size_t)l * Bn * CAND;
        float* boxes_l = boxes + (size_t)l * Bn * CAND * 4;
        float* areas_l = areas + (size_t)l * Bn * CAND;

        zero_k<<<(Bn * NBINS + 255) / 256, 256, 0, stream>>>(hist, Bn * NBINS);
        zero_k<<<1, 256, 0, stream>>>(cnt, Bn * 2);
        hist_k<<<dim3((N + 255) / 256, Bn), 256, 0, stream>>>(
            scores, lvl_off[l], TOTN, N, hist);
        thresh_k<<<Bn, 64, 0, stream>>>(hist, k, tinfo);
        compact_k<<<dim3((N + 255) / 256, Bn), 256, 0, stream>>>(
            scores, lvl_off[l], TOTN, N, k, tinfo, cnt, topv_l, topi_l);
        decode_k<<<dim3((k + 255) / 256, Bn), 256, 0, stream>>>(
            topi_l, reg, lvl_off[l], TOTN, Ws[l],
            (float)sizesA[l], (float)stridesA[l], k, ih, iw, boxes_l, areas_l);
        nms_k<<<Bn, 256, 0, stream>>>(boxes_l, areas_l, topv_l, k, n_out, l,
                                      catb, cats);
    }

    final_k<<<Bn, 256, 0, stream>>>(catb, cats, outf);
}
